// RefNet_encoder_15470472200810
// MI455X (gfx1250) — matrix operand order repaired and checked by reading, not compiled, not run on hardware
//
#include <hip/hip_runtime.h>
#include <math.h>

typedef __attribute__((ext_vector_type(16))) _Float16 v16h;
typedef __attribute__((ext_vector_type(8)))  float    v8f;

#define NBAT 4
#define CCH  64
#define WIMG 112
#define HWPX 12544       // 112*112
#define RWIN 7
#define RNUM 49          // 7*7 regions
#define RS   256         // 16*16 tokens per region
#define PTIL 784         // 12544/16 pixel tiles per image

// ---------------------------------------------------------------------------
// WMMA helpers (gfx1250, wave32): D = A(16x32 f16) * B(32x16 f16) + C(f32)
// ---------------------------------------------------------------------------
__device__ inline v8f wmma32(v16h a, v16h b, v8f c) {
  return __builtin_amdgcn_wmma_f32_16x16x32_f16(false, a, false, b, (short)0, c,
                                                false, false);
}

// A-fragment (16x32, f16). Lane l holds row m=l&15; element e maps to
// k = (v&3)*2 + (v>>2)*16 + (l>>4)*8 + (e&1) with v=e/2 (ISA 7.12.2 table).
template <typename F>
__device__ inline v16h frag_a16(int lane, F f) {
  v16h a;
  const int m = lane & 15, g = (lane >> 4) & 1;
#pragma unroll
  for (int e = 0; e < 16; ++e) {
    const int v = e >> 1;
    const int k = ((v & 3) << 1) + ((v >> 2) << 4) + (g << 3) + (e & 1);
    a[e] = (_Float16)f(m, k);
  }
  return a;
}

// B-fragment (32x16, f16). Lane l holds column n=l&15; element e maps to
// k = (e&7) + (e>>3)*16 + g*8 (same k order as frag_a16).
template <typename F>
__device__ inline v16h frag_b16(int lane, F f) {
  v16h b;
  const int n = lane & 15, g = (lane >> 4) & 1;
#pragma unroll
  for (int e = 0; e < 16; ++e) b[e] = (_Float16)f((e & 7) + ((e >> 3) << 4) + (g << 3), n);
  return b;
}

__device__ inline float grp16_max(float v) {
#pragma unroll
  for (int m = 8; m >= 1; m >>= 1) v = fmaxf(v, __shfl_xor(v, m, 32));
  return v;
}
__device__ inline float grp16_sum(float v) {
#pragma unroll
  for (int m = 8; m >= 1; m >>= 1) v += __shfl_xor(v, m, 32);
  return v;
}

// ---------------------------------------------------------------------------
// Async global -> LDS copy (CDNA5, tracked by ASYNCcnt).  lds_off is the
// byte offset within the workgroup LDS allocation (low 32 bits of the flat
// shared address); gaddr is the per-lane 64-bit global address.
// ---------------------------------------------------------------------------
__device__ inline void async_load_b128(unsigned lds_off, const void* gaddr) {
  asm volatile("global_load_async_to_lds_b128 %0, %1, off"
               :
               : "v"(lds_off), "v"((unsigned long long)(uintptr_t)gaddr)
               : "memory");
}
__device__ inline void wait_asynccnt0() {
  asm volatile("s_wait_asynccnt 0x0" ::: "memory");
}

// ---------------------------------------------------------------------------
// Bilinear resize (half-pixel centers, matches jax.image.resize bilinear)
// ---------------------------------------------------------------------------
__global__ void resize_kernel(const float* __restrict__ src,
                              float* __restrict__ dst, int NC, int inH, int inW,
                              int outH, int outW) {
  int gid = blockIdx.x * blockDim.x + threadIdx.x;
  int total = NC * outH * outW;
  if (gid >= total) return;
  int x = gid % outW, y = (gid / outW) % outH, nc = gid / (outW * outH);
  float sy = (y + 0.5f) * (float)inH / (float)outH - 0.5f;
  float sx = (x + 0.5f) * (float)inW / (float)outW - 0.5f;
  int y0 = (int)floorf(sy), x0 = (int)floorf(sx);
  float fy = sy - (float)y0, fx = sx - (float)x0;
  int y1 = min(y0 + 1, inH - 1), x1 = min(x0 + 1, inW - 1);
  y0 = max(y0, 0); x0 = max(x0, 0);
  const float* s = src + (size_t)nc * inH * inW;
  float v00 = s[y0 * inW + x0], v01 = s[y0 * inW + x1];
  float v10 = s[y1 * inW + x0], v11 = s[y1 * inW + x1];
  dst[gid] = v00 * (1 - fy) * (1 - fx) + v01 * (1 - fy) * fx +
             v10 * fy * (1 - fx) + v11 * fy * fx;
}

// sigmoid(pre)>0.5  <=>  pre>0 ;  attn_mask = m==0 ? -100 : 0
__global__ void mask_kernel(const float* __restrict__ pre,
                            float* __restrict__ m, int total) {
  int gid = blockIdx.x * blockDim.x + threadIdx.x;
  if (gid < total) m[gid] = (pre[gid] > 0.0f) ? 0.0f : -100.0f;
}

// ---------------------------------------------------------------------------
// conv0: 3x3, 131 -> 64 channels, SAME padding. Input gathered from the
// three concatenated sources without materializing the concat.
// ---------------------------------------------------------------------------
__global__ void conv0_kernel(const float* __restrict__ feats,
                             const float* __restrict__ exem,
                             const float* __restrict__ img,
                             const float* __restrict__ w,
                             const float* __restrict__ b,
                             float* __restrict__ out) {
  int gid = blockIdx.x * blockDim.x + threadIdx.x;
  if (gid >= NBAT * CCH * HWPX) return;
  int x = gid % WIMG, y = (gid / WIMG) % WIMG;
  int o = (gid / HWPX) % CCH, n = gid / (HWPX * CCH);
  float acc = b[o];
  for (int ci = 0; ci < 131; ++ci) {
    const float* src;
    if (ci < 64)        src = feats + ((size_t)n * 64 + ci) * HWPX;
    else if (ci < 128)  src = exem  + ((size_t)n * 64 + (ci - 64)) * HWPX;
    else                src = img   + ((size_t)n * 3  + (ci - 128)) * HWPX;
    const float* wr = w + ((size_t)o * 131 + ci) * 9;
    for (int ky = 0; ky < 3; ++ky) {
      int yy = y + ky - 1;
      if (yy < 0 || yy >= WIMG) continue;
      for (int kx = 0; kx < 3; ++kx) {
        int xx = x + kx - 1;
        if (xx < 0 || xx >= WIMG) continue;
        acc += wr[ky * 3 + kx] * src[yy * WIMG + xx];
      }
    }
  }
  out[gid] = acc;
}

// ---------------------------------------------------------------------------
// LayerNorm over channels (C=64) per pixel
// ---------------------------------------------------------------------------
__global__ void ln_kernel(const float* __restrict__ x,
                          const float* __restrict__ gg,
                          const float* __restrict__ bb,
                          float* __restrict__ out) {
  int gid = blockIdx.x * blockDim.x + threadIdx.x;
  if (gid >= NBAT * HWPX) return;
  int n = gid / HWPX, p = gid % HWPX;
  const float* xb = x + (size_t)n * CCH * HWPX + p;
  float s = 0.f, s2 = 0.f;
  for (int c = 0; c < CCH; ++c) {
    float v = xb[(size_t)c * HWPX];
    s += v; s2 += v * v;
  }
  float mu = s / CCH;
  float var = s2 / CCH - mu * mu;
  float inv = rsqrtf(var + 1e-5f);
  float* ob = out + (size_t)n * CCH * HWPX + p;
  for (int c = 0; c < CCH; ++c)
    ob[(size_t)c * HWPX] = (xb[(size_t)c * HWPX] - mu) * inv * gg[c] + bb[c];
}

// ---------------------------------------------------------------------------
// 1x1 conv as WMMA GEMM: out[n,o,p] = sum_c w[o,c]*x[n,c,p] + bias[o]
// One wave computes a 16(out-ch) x 16(pixel) tile; K tiled by 32.
// act: 0=none, 1=exact GELU. res: optional residual added in epilogue.
// ---------------------------------------------------------------------------
__global__ __launch_bounds__(256) void conv1x1_wmma(
    const float* __restrict__ x, const float* __restrict__ wgt,
    const float* __restrict__ bias, const float* __restrict__ res,
    float* __restrict__ out, int Cin, int Cout, int act) {
  const int wid = blockIdx.x * 8 + (threadIdx.x >> 5);
  const int lane = threadIdx.x & 31;
  const int mtiles = Cout >> 4;
  const int total = NBAT * mtiles * PTIL;
  if (wid >= total) return;
  const int mt = wid % mtiles;
  const int pt = (wid / mtiles) % PTIL;
  const int n  = wid / (mtiles * PTIL);
  const float* xb = x + (size_t)n * Cin * HWPX + pt * 16;

  v8f acc = v8f{};
  for (int k0 = 0; k0 < Cin; k0 += 32) {
    if (k0 + 32 < Cin)  // warm L2 for the next K panel (global_prefetch_b8)
      __builtin_prefetch(xb + (size_t)(k0 + 32 + (lane & 15)) * HWPX, 0, 1);
    v16h a = frag_a16(lane, [&](int m, int k) {
      return wgt[(size_t)(mt * 16 + m) * Cin + k0 + k];
    });
    v16h b = frag_b16(lane, [&](int k, int nn) {
      return xb[(size_t)(k0 + k) * HWPX + nn];
    });
    acc = wmma32(a, b, acc);
  }
  const int col = pt * 16 + (lane & 15);
  const int g = (lane >> 4) & 1;
#pragma unroll
  for (int e = 0; e < 8; ++e) {
    const int o = mt * 16 + e + (g << 3);
    float v = acc[e] + bias[o];
    if (act == 1) v = 0.5f * v * (1.0f + erff(v * 0.70710678118f));
    const size_t oi = ((size_t)n * Cout + o) * HWPX + col;
    if (res) v += res[oi];
    out[oi] = v;
  }
}

// ---------------------------------------------------------------------------
// Region mean pooling of q (ch 0..63) and k (ch 64..127)
// ---------------------------------------------------------------------------
__global__ void pool_kernel(const float* __restrict__ qkv,
                            float* __restrict__ qd, float* __restrict__ kd) {
  int gid = blockIdx.x * blockDim.x + threadIdx.x;
  if (gid >= NBAT * CCH * RNUM) return;
  int p = gid % RNUM, c = (gid / RNUM) % CCH, n = gid / (RNUM * CCH);
  int wy = p / RWIN, wx = p % RWIN;
  const float* qb = qkv + ((size_t)n * 192 + c) * HWPX;
  const float* kb = qkv + ((size_t)n * 192 + 64 + c) * HWPX;
  float sq = 0.f, sk = 0.f;
  for (int iy = 0; iy < 16; ++iy)
    for (int ix = 0; ix < 16; ++ix) {
      int pix = (wy * 16 + iy) * WIMG + wx * 16 + ix;
      sq += qb[pix]; sk += kb[pix];
    }
  qd[gid] = sq * (1.0f / 256.0f);
  kd[gid] = sk * (1.0f / 256.0f);
}

// ---------------------------------------------------------------------------
// Region affinity a_r[p,q'] = sum_c qd[c,p]*kd[c,q'] and stable top-4
// ---------------------------------------------------------------------------
__global__ void topk_kernel(const float* __restrict__ qd,
                            const float* __restrict__ kd,
                            int* __restrict__ idx) {
  int n = blockIdx.x / RNUM, p = blockIdx.x % RNUM;
  __shared__ float ar[RNUM];
  int t = threadIdx.x;
  if (t < RNUM) {
    float s = 0.f;
    for (int c = 0; c < CCH; ++c)
      s += qd[((size_t)n * CCH + c) * RNUM + p] *
           kd[((size_t)n * CCH + c) * RNUM + t];
    ar[t] = s;
  }
  __syncthreads();
  if (t == 0) {
    bool used[RNUM] = {false};
    for (int kk = 0; kk < 4; ++kk) {
      int best = 0; float bv = -INFINITY;
      for (int j = 0; j < RNUM; ++j)
        if (!used[j] && ar[j] > bv) { bv = ar[j]; best = j; }
      used[best] = true;
      idx[((size_t)n * RNUM + p) * 4 + kk] = best;
    }
  }
}

// ---------------------------------------------------------------------------
// Routing attention: one block per (n, region). 8 waves x 32 query rows.
// Gathered K and V regions are staged in LDS with ASYNC global->LDS DMA
// (global_load_async_to_lds_b128 / s_wait_asynccnt). Flash-style online
// softmax; S = QK^T and O = PV via v_wmma_f32_16x16x32_f16.
// LDS: sKf 64K + sVf 64K + sP 16K + misc 3K = 147 KB (<= 2 WGs per 320KB WGP)
// ---------------------------------------------------------------------------
__global__ __launch_bounds__(256) void attn_kernel(
    const float* __restrict__ qkv, const float* __restrict__ amask,
    const int* __restrict__ idx, float* __restrict__ outp) {
  const int n = blockIdx.x / RNUM;
  const int p = blockIdx.x % RNUM;
  const int tid = threadIdx.x;
  const int w = tid >> 5;
  const int lane = tid & 31;
  const int g = (lane >> 4) & 1;

  __shared__ float    sKf[CCH * RS];     // 64 KB, [channel][token]
  __shared__ float    sVf[CCH * RS];     // 64 KB, [channel][token]
  __shared__ float    sMask[RS];
  __shared__ float    sM[8][32], sL[8][32];
  __shared__ _Float16 sP[8][32 * 32];    // 16 KB: per-wave P tile scratch

  int ridx[4];
#pragma unroll
  for (int t = 0; t < 4; ++t) ridx[t] = idx[((size_t)n * RNUM + p) * 4 + t];

  sM[tid >> 5][tid & 31] = -INFINITY;
  sL[tid >> 5][tid & 31] = 0.0f;

  const int wy = p / RWIN, wx = p % RWIN;
  const float scale = 0.125f;  // hd^-0.5, hd=64
  const float* qbase = qkv + (size_t)n * 192 * HWPX;
  const float* kgbase = qbase + (size_t)64 * HWPX;
  const float* vgbase = qbase + (size_t)128 * HWPX;
  const unsigned ldsK = (unsigned)(uintptr_t)(void*)sKf;
  const unsigned ldsV = (unsigned)(uintptr_t)(void*)sVf;

  // Q fragments for this wave's 32 query rows (pre-scaled)
  v16h qf[2][2];
#pragma unroll
  for (int mt = 0; mt < 2; ++mt)
#pragma unroll
    for (int ks = 0; ks < 2; ++ks)
      qf[mt][ks] = frag_a16(lane, [&](int m, int k) {
        int tok = w * 32 + mt * 16 + m;
        int pix = (wy * 16 + (tok >> 4)) * WIMG + wx * 16 + (tok & 15);
        return qbase[(size_t)(ks * 32 + k) * HWPX + pix] * scale;
      });

  v8f O[2][4];
#pragma unroll
  for (int mt = 0; mt < 2; ++mt)
#pragma unroll
    for (int dt = 0; dt < 4; ++dt) O[mt][dt] = v8f{};

  __syncthreads();

#pragma unroll 1
  for (int t = 0; t < 4; ++t) {
    const int rg = ridx[t];
    const int ry = rg / RWIN, rx = rg % RWIN;

    // Async-DMA the gathered K/V region into LDS. Each 16-byte segment is
    // 4 consecutive pixels of one channel: contiguous in global (region
    // rows are 16 contiguous pixels) and contiguous in [ch][tok] LDS.
#pragma unroll 1
    for (int s = tid; s < 4096; s += 256) {
      int c = s >> 6, r = (s >> 2) & 15, q = s & 3;
      unsigned loff = (unsigned)(((c << 8) + (r << 4) + (q << 2)) << 2);
      int pix = (ry * 16 + r) * WIMG + rx * 16 + q * 4;
      async_load_b128(ldsK + loff, kgbase + (size_t)c * HWPX + pix);
      async_load_b128(ldsV + loff, vgbase + (size_t)c * HWPX + pix);
    }
    {
      const int tok = tid;
      const int pix = (ry * 16 + (tok >> 4)) * WIMG + rx * 16 + (tok & 15);
      sMask[tok] = amask[(size_t)n * HWPX + pix];
    }
    wait_asynccnt0();
    __syncthreads();

    // Warm L2 for the next gathered region while computing this one.
    if (t < 3) {
      const int rg2 = ridx[t + 1];
      const int py = (rg2 / RWIN) * 16 + ((tid >> 6) << 2);
      const int px = (rg2 % RWIN) * 16;
      const size_t lo = (size_t)(tid & 63) * HWPX + py * WIMG + px;
      __builtin_prefetch(kgbase + lo, 0, 1);
      __builtin_prefetch(vgbase + lo, 0, 1);
    }

#pragma unroll 1
    for (int j0 = 0; j0 < RS; j0 += 32) {
#pragma unroll 1
      for (int mt = 0; mt < 2; ++mt) {
        v8f Sa = v8f{}, Sb = v8f{};
#pragma unroll
        for (int ks = 0; ks < 2; ++ks) {
          v16h ba = frag_b16(lane, [&](int k, int nn) {
            return sKf[(ks * 32 + k) * RS + j0 + nn];
          });
          v16h bb = frag_b16(lane, [&](int k, int nn) {
            return sKf[(ks * 32 + k) * RS + j0 + 16 + nn];
          });
          Sa = wmma32(qf[mt][ks], ba, Sa);
          Sb = wmma32(qf[mt][ks], bb, Sb);
        }
        const float ma = sMask[j0 + (lane & 15)];
        const float mb = sMask[j0 + 16 + (lane & 15)];
#pragma unroll
        for (int e = 0; e < 8; ++e) {
          float sa = Sa[e] + ma, sb = Sb[e] + mb;
          float tmax = grp16_max(fmaxf(sa, sb));
          const int row = mt * 16 + e + (g << 3);
          float old_m = sM[w][row];
          float new_m = fmaxf(old_m, tmax);
          float corr = __expf(old_m - new_m);
          float pa = __expf(sa - new_m);
          float pb = __expf(sb - new_m);
          float psum = grp16_sum(pa + pb);
          if ((lane & 15) == 0) {
            sM[w][row] = new_m;
            sL[w][row] = sL[w][row] * corr + psum;
          }
#pragma unroll
          for (int dt = 0; dt < 4; ++dt) O[mt][dt][e] *= corr;
          sP[w][row * 32 + (lane & 15)] = (_Float16)pa;
          sP[w][row * 32 + 16 + (lane & 15)] = (_Float16)pb;
        }
      }
      // P tiles (C-layout -> A-layout via per-wave LDS scratch)
      v16h pf0 = frag_a16(lane, [&](int m, int k) {
        return (float)sP[w][m * 32 + k];
      });
      v16h pf1 = frag_a16(lane, [&](int m, int k) {
        return (float)sP[w][(16 + m) * 32 + k];
      });
#pragma unroll 1
      for (int dt = 0; dt < 4; ++dt) {
        v16h vf = frag_b16(lane, [&](int k, int nn) {
          return sVf[(dt * 16 + nn) * RS + j0 + k];
        });
        O[0][dt] = wmma32(pf0, vf, O[0][dt]);
        O[1][dt] = wmma32(pf1, vf, O[1][dt]);
      }
    }
    __syncthreads();
  }

  // epilogue: divide by row sum, scatter back to NCHW (== _from_regions)
#pragma unroll
  for (int mt = 0; mt < 2; ++mt) {
#pragma unroll
    for (int e = 0; e < 8; ++e) {
      const int row = mt * 16 + e + (g << 3);
      const float li = sL[w][row];
      const int tok = w * 32 + row;
      const int pix = (wy * 16 + (tok >> 4)) * WIMG + wx * 16 + (tok & 15);
#pragma unroll
      for (int dt = 0; dt < 4; ++dt) {
        const int d = dt * 16 + (lane & 15);
        outp[((size_t)n * CCH + d) * HWPX + pix] = O[mt][dt][e] / li;
      }
    }
  }
}

// ---------------------------------------------------------------------------
// LEPE: 5x5 depthwise conv of v (qkv ch 128..191) + bias, added into attn out
// ---------------------------------------------------------------------------
__global__ void lepe_kernel(const float* __restrict__ qkv,
                            const float* __restrict__ lw,
                            const float* __restrict__ lb,
                            float* __restrict__ outp) {
  int gid = blockIdx.x * blockDim.x + threadIdx.x;
  if (gid >= NBAT * CCH * HWPX) return;
  int x = gid % WIMG, y = (gid / WIMG) % WIMG;
  int c = (gid / HWPX) % CCH, n = gid / (HWPX * CCH);
  const float* v = qkv + ((size_t)n * 192 + 128 + c) * HWPX;
  float acc = lb[c];
  for (int ky = 0; ky < 5; ++ky) {
    int yy = y + ky - 2;
    if (yy < 0 || yy >= WIMG) continue;
    for (int kx = 0; kx < 5; ++kx) {
      int xx = x + kx - 2;
      if (xx < 0 || xx >= WIMG) continue;
      acc += lw[c * 25 + ky * 5 + kx] * v[yy * WIMG + xx];
    }
  }
  outp[gid] += acc;
}

// ---------------------------------------------------------------------------
// Host launch
// ---------------------------------------------------------------------------
extern "C" void kernel_launch(void* const* d_in, const int* in_sizes, int n_in,
                              void* d_out, int out_size, void* d_ws,
                              size_t ws_size, hipStream_t stream) {
  const float* exemplar0    = (const float*)d_in[0];
  const float* exemplar     = (const float*)d_in[1];
  const float* input1       = (const float*)d_in[2];
  const float* exemplar_pre = (const float*)d_in[3];
  const float* conv0_w = (const float*)d_in[4];
  const float* conv0_b = (const float*)d_in[5];
  const float* ln1_g = (const float*)d_in[6];
  const float* ln1_b = (const float*)d_in[7];
  const float* ln2_g = (const float*)d_in[8];
  const float* ln2_b = (const float*)d_in[9];
  const float* qkv_w = (const float*)d_in[10];
  const float* qkv_b = (const float*)d_in[11];
  const float* lepe_w = (const float*)d_in[12];
  const float* lepe_b = (const float*)d_in[13];
  const float* proj_w = (const float*)d_in[14];
  const float* proj_b = (const float*)d_in[15];
  const float* fc1_w = (const float*)d_in[16];
  const float* fc1_b = (const float*)d_in[17];
  const float* fc2_w = (const float*)d_in[18];
  const float* fc2_b = (const float*)d_in[19];

  float* hx = (float*)d_out;  // (4,64,112,112)

  // workspace carve
  float* ws = (float*)d_ws;
  size_t off = 0;
  auto alloc = [&](size_t nelem) { float* pp = ws + off; off += nelem; return pp; };
  float* feats_s = alloc((size_t)NBAT * 64 * HWPX);
  float* img_s   = alloc((size_t)NBAT * 3 * HWPX);
  float* pre_s   = alloc((size_t)NBAT * HWPX);
  float* amask   = alloc((size_t)NBAT * HWPX);
  float* xln     = alloc((size_t)NBAT * 64 * HWPX);
  float* qkvb    = alloc((size_t)NBAT * 192 * HWPX);
  float* attn_o  = alloc((size_t)NBAT * 64 * HWPX);
  float* hbuf    = alloc((size_t)NBAT * 256 * HWPX);
  float* qd      = alloc((size_t)NBAT * 64 * RNUM);
  float* kd      = alloc((size_t)NBAT * 64 * RNUM);
  int*   idxb    = (int*)alloc((size_t)NBAT * RNUM * 4);

  const int TB = 256;
  auto gr = [](size_t n, int tb) { return (int)((n + tb - 1) / tb); };

  // resizes 224->112
  resize_kernel<<<gr((size_t)NBAT * 64 * HWPX, TB), TB, 0, stream>>>(
      exemplar0, feats_s, NBAT * 64, 224, 224, 112, 112);
  resize_kernel<<<gr((size_t)NBAT * 3 * HWPX, TB), TB, 0, stream>>>(
      input1, img_s, NBAT * 3, 224, 224, 112, 112);
  resize_kernel<<<gr((size_t)NBAT * HWPX, TB), TB, 0, stream>>>(
      exemplar_pre, pre_s, NBAT, 224, 224, 112, 112);
  mask_kernel<<<gr((size_t)NBAT * HWPX, TB), TB, 0, stream>>>(
      pre_s, amask, NBAT * HWPX);

  conv0_kernel<<<gr((size_t)NBAT * CCH * HWPX, TB), TB, 0, stream>>>(
      feats_s, exemplar, img_s, conv0_w, conv0_b, hx);

  for (int i = 0; i < 4; ++i) {
    // --- attention branch ---
    ln_kernel<<<gr((size_t)NBAT * HWPX, TB), TB, 0, stream>>>(
        hx, ln1_g + i * 64, ln1_b + i * 64, xln);
    conv1x1_wmma<<<392 * 12, TB, 0, stream>>>(
        xln, qkv_w + (size_t)i * 192 * 64, qkv_b + i * 192, nullptr, qkvb,
        64, 192, 0);
    pool_kernel<<<gr((size_t)NBAT * CCH * RNUM, TB), TB, 0, stream>>>(
        qkvb, qd, kd);
    topk_kernel<<<NBAT * RNUM, 64, 0, stream>>>(qd, kd, idxb);
    attn_kernel<<<NBAT * RNUM, 256, 0, stream>>>(qkvb, amask, idxb, attn_o);
    lepe_kernel<<<gr((size_t)NBAT * CCH * HWPX, TB), TB, 0, stream>>>(
        qkvb, lepe_w + (size_t)i * 64 * 25, lepe_b + i * 64, attn_o);
    conv1x1_wmma<<<392 * 4, TB, 0, stream>>>(
        attn_o, proj_w + (size_t)i * 64 * 64, proj_b + i * 64, hx, hx,
        64, 64, 0);

    // --- MLP branch ---
    ln_kernel<<<gr((size_t)NBAT * HWPX, TB), TB, 0, stream>>>(
        hx, ln2_g + i * 64, ln2_b + i * 64, xln);
    conv1x1_wmma<<<392 * 16, TB, 0, stream>>>(
        xln, fc1_w + (size_t)i * 256 * 64, fc1_b + i * 256, nullptr, hbuf,
        64, 256, 1);
    conv1x1_wmma<<<392 * 4, TB, 0, stream>>>(
        hbuf, fc2_w + (size_t)i * 64 * 256, fc2_b + i * 64, hx, hx,
        256, 64, 0);
  }
}
